// KalmanFormerNet_37185826848902
// MI455X (gfx1250) — compile-verified, matrix operand order checked by reading, not run on hardware
//
#include <hip/hip_runtime.h>
#include <hip/hip_bf16.h>

typedef __attribute__((ext_vector_type(16))) _Float16 v16h;
typedef __attribute__((ext_vector_type(8)))  float    v8f;
typedef _Float16 h16;

#define DEV __device__ __forceinline__

constexpr int Bc = 16, Tc = 64, Nc = 196, Dc = 128;
constexpr int NPAD = 208;      // tokens padded to 13*16
constexpr int SPAD = 224;      // attention K padded to 7*32
constexpr int ND   = Nc * Dc;  // 25088
constexpr float THRESHV = 0.005f;

// ---------------------------------------------------------------------------
// CDNA5 WMMA 16-bit fragment index helpers (CDNA5 ISA §7.12.2)
// A 16x32 f16 tile: lane = hi*16 + row;  half j -> K index:
//   vgpr 0..3 -> K = kbase + 2*vgpr + (j&1), vgpr 4..7 -> K = kbase+16+...
// ---------------------------------------------------------------------------
DEV int kmap(int j, int hi) {
  int vg = j >> 1, j1 = j & 1;
  int base = hi ? 8 : 0;
  int k = (vg < 4) ? (base + 2 * vg) : (base + 16 + 2 * (vg - 4));
  return k + j1;
}

// A fragment where row K-space is split across two LDS arrays (enc/dec concat)
DEV v16h loadA2(const h16* lo, const h16* hi, int rt, int kt, int lane) {
  int row = rt * 16 + (lane & 15);
  int hl  = (lane >> 4) & 1;
  v16h a;
#pragma unroll
  for (int j = 0; j < 16; ++j) {
    int kk = kt * 32 + kmap(j, hl);
    a[j] = (kk < Dc) ? lo[row * Dc + kk] : hi[row * Dc + (kk - Dc)];
  }
  return a;
}

DEV v16h loadA1(const h16* src, int stride, int rt, int kt, int lane) {
  int row = rt * 16 + (lane & 15);
  int hl  = (lane >> 4) & 1;
  v16h a;
#pragma unroll
  for (int j = 0; j < 16; ++j) a[j] = src[row * stride + kt * 32 + kmap(j, hl)];
  return a;
}

// B[k][n] where LDS holds src[n][k]  (q @ k^T: k stored [token][feature])
DEV v16h loadBT(const h16* src, int stride, int nt, int kt, int lane) {
  int col = nt * 16 + (lane & 15);
  int hl  = (lane >> 4) & 1;
  v16h b;
#pragma unroll
  for (int j = 0; j < 16; ++j) b[j] = src[col * stride + kt * 32 + kmap(j, hl)];
  return b;
}

// B[k][n] where LDS holds src[k][n]  (p @ v), guarded against K padding
DEV v16h loadBD(const h16* src, int stride, int nt, int kt, int lane, int maxK) {
  int col = nt * 16 + (lane & 15);
  int hl  = (lane >> 4) & 1;
  v16h b;
#pragma unroll
  for (int j = 0; j < 16; ++j) {
    int kk = kt * 32 + kmap(j, hl);
    b[j] = (kk < maxK) ? src[kk * stride + col] : (h16)0.0f;
  }
  return b;
}

// Weights pre-packed fragment-linear: one contiguous 32B load per lane
DEV v16h loadBW(const h16* wf, int NT, int kt, int nt, int lane) {
  return *(const v16h*)(wf + (((kt * NT + nt) * 32 + lane) << 4));
}

// C/D f32 16x16 layout: lane 0-15 -> N=lane, M=vgpr; lane 16-31 -> M=vgpr+8
DEV void storeD16(h16* dst, int stride, int rt, int nt, v8f acc, float scl, int lane) {
  int col   = nt * 16 + (lane & 15);
  int rbase = rt * 16 + ((lane >> 4) & 1) * 8;
#pragma unroll
  for (int v = 0; v < 8; ++v) dst[(rbase + v) * stride + col] = (h16)(acc[v] * scl);
}

DEV v8f wmma32f16(v16h a, v16h b, v8f c) {
  return __builtin_amdgcn_wmma_f32_16x16x32_f16(false, a, false, b, (short)0, c,
                                                false, false);
}

// ---------------------------------------------------------------------------
// Kernel 0: convert f32 weights into f16 WMMA-B fragment-linear layout
// out[((kt*NT+nt)*32+lane)*16+j] = (h16) W[(kt*32+kmap(j,lane>=16))*ncols + nt*16+(lane&15)]
// ---------------------------------------------------------------------------
__global__ void prep_weights(const float* __restrict__ W, h16* __restrict__ out,
                             int KT, int NT, int ncols) {
  int idx = blockIdx.x * blockDim.x + threadIdx.x;
  int total = KT * NT * 32 * 16;
  if (idx >= total) return;
  int j = idx & 15, lane = (idx >> 4) & 31, tile = idx >> 9;
  int nt = tile % NT, kt = tile / NT;
  int kk = kt * 32 + kmap(j, lane >> 4);
  int n  = nt * 16 + (lane & 15);
  out[idx] = (h16)W[kk * ncols + n];
}

// Kernel: zero scan state + emit cu_seqlen int32 tail
__global__ void init_state_cu(float* __restrict__ vel, float* __restrict__ innov,
                              float* __restrict__ sud, int n, int* __restrict__ cu) {
  int idx = blockIdx.x * blockDim.x + threadIdx.x;
  if (idx < n) { vel[idx] = 0.f; innov[idx] = 0.f; sud[idx] = 0.f; }
  if (idx <= Bc) cu[idx] = idx * (Tc * Nc);
}

// ---------------------------------------------------------------------------
// Kernel 1: temporal-attention token gating -> sel; also writes t==0 output
// ---------------------------------------------------------------------------
__global__ __launch_bounds__(256)
void token_select(const float* __restrict__ x, const float* __restrict__ w_attn,
                  float* __restrict__ sel, float* __restrict__ out) {
  int bt = blockIdx.x;
  int t  = bt % Tc;
  const float* xb = x + (size_t)bt * ND;
  __shared__ float score[Nc];
  __shared__ float s_mx, s_sm;
  int tid = threadIdx.x, lane = tid & 31, wave = tid >> 5;

  for (int n = wave; n < Nc; n += 8) {
    float p = 0.f;
    for (int d = lane; d < Dc; d += 32) p += xb[n * Dc + d] * w_attn[d];
#pragma unroll
    for (int m = 16; m; m >>= 1) p += __shfl_xor(p, m, 32);
    if (lane == 0) score[n] = p;
  }
  __syncthreads();
  if (wave == 0) {
    float mx = -1e30f;
    for (int n = lane; n < Nc; n += 32) mx = fmaxf(mx, score[n]);
#pragma unroll
    for (int m = 16; m; m >>= 1) mx = fmaxf(mx, __shfl_xor(mx, m, 32));
    float sm = 0.f;
    for (int n = lane; n < Nc; n += 32) sm += __expf(score[n] - mx);
#pragma unroll
    for (int m = 16; m; m >>= 1) sm += __shfl_xor(sm, m, 32);
    if (lane == 0) { s_mx = mx; s_sm = sm; }
  }
  __syncthreads();
  float inv = 1.f / s_sm, mx = s_mx;
  for (int i = tid; i < ND; i += 256) {
    int n = i >> 7;
    float p = __expf(score[n] - mx) * inv;
    float v = xb[i];
    float o = (p > THRESHV) ? v : 0.f;
    sel[(size_t)bt * ND + i] = o;
    if (t == 0) out[(size_t)bt * ND + i] = o;  // x_post_all[:,0] = sel[:,0]
  }
}

// ---------------------------------------------------------------------------
// Kernel 2: sequential Kalman scan. One block per batch, 512 threads (16 waves).
// All per-step matrices live in LDS (f16), regions recycled phase-by-phase.
// LDS half-offsets (each [208][128] = 26624 halves):
//   0      innov        (enc lo)   | reused as softmax buffer s [208][224]
//   26624  vel          (enc hi)   |   (s = 46592 halves, fits in first two)
//   53248  dinnov -> q
//   79872  psud   -> a@v
//   106496 k      -> K-gain
//   133120 v
// total 159744 halves = 319488 B  (< 320 KB WGP LDS)
// ---------------------------------------------------------------------------
__global__ __launch_bounds__(512)
void kalman_scan(const float* __restrict__ sel,
                 float* __restrict__ g_vel, float* __restrict__ g_innov,
                 float* __restrict__ g_sud,
                 const h16* __restrict__ wq, const h16* __restrict__ wk,
                 const h16* __restrict__ wv, const h16* __restrict__ wo,
                 float* __restrict__ out) {
  extern __shared__ char smemRaw[];
  h16* smem   = (h16*)smemRaw;
  h16* L_innov = smem;            // enc lo
  h16* L_vel   = smem + 26624;    // enc hi
  h16* L_q     = smem + 53248;    // dec lo (dinnov), then q
  h16* L_av    = smem + 79872;    // dec hi (psud),  then a@v
  h16* L_kg    = smem + 106496;   // k, then K-gain
  h16* L_v     = smem + 133120;   // v
  h16* L_s     = smem;            // [208][224] scores/probs (over enc regions)

  int b = blockIdx.x;
  int tid = threadIdx.x, lane = tid & 31, wave = tid >> 5;
  const float* selB = sel + (size_t)b * Tc * ND;
  float* vB = g_vel   + (size_t)b * ND;
  float* iB = g_innov + (size_t)b * ND;
  float* sB = g_sud   + (size_t)b * ND;
  float* outB = out + (size_t)b * Tc * ND;
  const float qscale = 0.08838834764831845f;  // 1/sqrt(128)

  for (int t = 1; t < Tc; ++t) {
    const float* prevA = selB + (size_t)(t - 1) * ND;
    const float* obsA  = selB + (size_t)t * ND;

    // --- Phase A: build f16 enc/dec tiles, zero pad rows 196..207 ---
    for (int i = tid; i < NPAD * Dc; i += 512) {
      int row = i >> 7;
      if (row < Nc) {
        float velv = vB[i];
        float pin  = iB[i];
        float psd  = sB[i];
        float pred = prevA[i] + velv;
        float innv = obsA[i] - pred;
        L_innov[i] = (h16)innv;
        L_vel[i]   = (h16)velv;
        L_q[i]     = (h16)(innv - pin);  // dinnov
        L_av[i]    = (h16)psd;           // prev sud
      } else {
        L_innov[i] = (h16)0.f; L_vel[i] = (h16)0.f;
        L_q[i] = (h16)0.f;     L_av[i] = (h16)0.f;
      }
    }
    __syncthreads();

    // --- Phase B: k = dec@Wk, v = dec@Wv (13 row-tiles x 8 n-tiles x 2) ---
    for (int task = wave; task < 13 * 8 * 2; task += 16) {
      int which = task & 1, rt = (task >> 1) / 8, nt = (task >> 1) & 7;
      v8f acc = {};
#pragma unroll
      for (int kt = 0; kt < 8; ++kt) {
        v16h a  = loadA2(L_q, L_av, rt, kt, lane);
        v16h bb = loadBW(which ? wv : wk, 8, kt, nt, lane);
        acc = wmma32f16(a, bb, acc);
      }
      storeD16(which ? L_v : L_kg, Dc, rt, nt, acc, 1.0f, lane);
    }
    __syncthreads();

    // --- Phase C: q = enc@Wq (overwrites dinnov region) ---
    for (int task = wave; task < 13 * 8; task += 16) {
      int rt = task / 8, nt = task & 7;
      v8f acc = {};
#pragma unroll
      for (int kt = 0; kt < 8; ++kt) {
        v16h a  = loadA2(L_innov, L_vel, rt, kt, lane);
        v16h bb = loadBW(wq, 8, kt, nt, lane);
        acc = wmma32f16(a, bb, acc);
      }
      storeD16(L_q, Dc, rt, nt, acc, 1.0f, lane);
    }
    __syncthreads();

    // --- Phase D: s = scale * q @ k^T  (f16 scores [208][224]) ---
    // zero pad columns 208..223 (disjoint from tile writes below)
    for (int i = tid; i < NPAD * 16; i += 512) {
      int r = i >> 4, c = (SPAD - 16) + (i & 15);
      L_s[r * SPAD + c] = (h16)0.f;
    }
    for (int task = wave; task < 13 * 13; task += 16) {
      int rt = task / 13, mt = task % 13;
      v8f acc = {};
#pragma unroll
      for (int kt = 0; kt < 4; ++kt) {
        v16h a  = loadA1(L_q, Dc, rt, kt, lane);
        v16h bb = loadBT(L_kg, Dc, mt, kt, lane);  // k is [token][feature]
        acc = wmma32f16(a, bb, acc);
      }
      storeD16(L_s, SPAD, rt, mt, acc, qscale, lane);
    }
    __syncthreads();

    // --- Phase E1: row softmax over 196 columns (one wave per row) ---
    for (int r = wave; r < Nc; r += 16) {
      h16* srow = L_s + r * SPAD;
      float mx = -1e30f;
      for (int c = lane; c < Nc; c += 32) mx = fmaxf(mx, (float)srow[c]);
#pragma unroll
      for (int m = 16; m; m >>= 1) mx = fmaxf(mx, __shfl_xor(mx, m, 32));
      float sm = 0.f;
      for (int c = lane; c < Nc; c += 32) sm += __expf((float)srow[c] - mx);
#pragma unroll
      for (int m = 16; m; m >>= 1) sm += __shfl_xor(sm, m, 32);
      float inv = 1.f / sm;
      for (int c = lane; c < Nc; c += 32)
        srow[c] = (h16)(__expf((float)srow[c] - mx) * inv);
    }
    __syncthreads();

    // --- Phase E2: a@v = p @ v  (K = 224 with guard; v pad rows are 0) ---
    for (int task = wave; task < 13 * 8; task += 16) {
      int rt = task / 8, nt = task & 7;
      v8f acc = {};
#pragma unroll
      for (int kt = 0; kt < 7; ++kt) {
        v16h a  = loadA1(L_s, SPAD, rt, kt, lane);
        v16h bb = loadBD(L_v, Dc, nt, kt, lane, NPAD);
        acc = wmma32f16(a, bb, acc);
      }
      storeD16(L_av, Dc, rt, nt, acc, 1.0f, lane);
    }
    __syncthreads();

    // --- Phase F: K-gain = (a@v) @ Wo  -> L_kg (k region is dead) ---
    for (int task = wave; task < 13 * 8; task += 16) {
      int rt = task / 8, nt = task & 7;
      v8f acc = {};
#pragma unroll
      for (int kt = 0; kt < 4; ++kt) {
        v16h a  = loadA1(L_av, Dc, rt, kt, lane);
        v16h bb = loadBW(wo, 8, kt, nt, lane);
        acc = wmma32f16(a, bb, acc);
      }
      storeD16(L_kg, Dc, rt, nt, acc, 1.0f, lane);
    }
    __syncthreads();

    // --- Phase G: posterior update + state write-back ---
    for (int i = tid; i < ND; i += 512) {
      float velv = vB[i];
      float pred = prevA[i] + velv;
      float innv = obsA[i] - pred;
      float dx   = (float)L_kg[i] * innv;
      outB[(size_t)t * ND + i] = pred + dx;   // x_post
      vB[i] = velv + dx;                      // new vel = vel + K*innov
      iB[i] = innv;                           // prev innov
      sB[i] = dx;                             // prev sud = K*innov
      if (t + 1 < Tc) __builtin_prefetch(obsA + ND + i, 0, 0);
    }
    __threadfence_block();
    __syncthreads();
  }
}

// ---------------------------------------------------------------------------
extern "C" void kernel_launch(void* const* d_in, const int* in_sizes, int n_in,
                              void* d_out, int out_size, void* d_ws, size_t ws_size,
                              hipStream_t stream) {
  (void)in_sizes; (void)n_in; (void)out_size; (void)ws_size;
  const float* x  = (const float*)d_in[0];
  const float* Wq = (const float*)d_in[1];
  const float* Wk = (const float*)d_in[2];
  const float* Wv = (const float*)d_in[3];
  const float* Wo = (const float*)d_in[4];
  const float* wa = (const float*)d_in[5];
  float* out = (float*)d_out;

  // workspace carve-up
  char* ws = (char*)d_ws;
  size_t off = 0;
  float* sel   = (float*)(ws + off); off += (size_t)Bc * Tc * ND * 4;
  float* vel   = (float*)(ws + off); off += (size_t)Bc * ND * 4;
  float* innov = (float*)(ws + off); off += (size_t)Bc * ND * 4;
  float* sud   = (float*)(ws + off); off += (size_t)Bc * ND * 4;
  h16* wqf = (h16*)(ws + off); off += (size_t)8 * 8 * 32 * 16 * 2;
  h16* wkf = (h16*)(ws + off); off += (size_t)8 * 8 * 32 * 16 * 2;
  h16* wvf = (h16*)(ws + off); off += (size_t)8 * 8 * 32 * 16 * 2;
  h16* wof = (h16*)(ws + off); off += (size_t)4 * 8 * 32 * 16 * 2;

  // f32 weights -> f16 fragment-linear (L2-resident, reused for all 63 steps)
  prep_weights<<<(8 * 8 * 32 * 16 + 255) / 256, 256, 0, stream>>>(Wq, wqf, 8, 8, Dc);
  prep_weights<<<(8 * 8 * 32 * 16 + 255) / 256, 256, 0, stream>>>(Wk, wkf, 8, 8, Dc);
  prep_weights<<<(8 * 8 * 32 * 16 + 255) / 256, 256, 0, stream>>>(Wv, wvf, 8, 8, Dc);
  prep_weights<<<(4 * 8 * 32 * 16 + 255) / 256, 256, 0, stream>>>(Wo, wof, 4, 8, Dc);

  int nstate = Bc * ND;
  init_state_cu<<<(nstate + 255) / 256, 256, 0, stream>>>(
      vel, innov, sud, nstate, (int*)out + (size_t)Bc * Tc * ND);

  token_select<<<Bc * Tc, 256, 0, stream>>>(x, wa, sel, out);

  constexpr int SMEM_BYTES = 159744 * 2;  // 319488 B < 320 KB WGP LDS
  (void)hipFuncSetAttribute(reinterpret_cast<const void*>(kalman_scan),
                            hipFuncAttributeMaxDynamicSharedMemorySize, SMEM_BYTES);
  kalman_scan<<<Bc, 512, SMEM_BYTES, stream>>>(sel, vel, innov, sud,
                                               wqf, wkf, wvf, wof, out);
}
